// GeneratorNetwork_6468220748211
// MI455X (gfx1250) — compile-verified, matrix operand order checked by reading, not run on hardware
//
#include <hip/hip_runtime.h>
#include <hip/hip_bf16.h>
#include <math.h>

// ---------------------------------------------------------------------------
// Problem constants (from reference)
// ---------------------------------------------------------------------------
#define BSZ  256   // batch
#define LAT  256
#define CODE 2048
#define RNN  512
#define G4   (4*RNN)   // 2048 gate width
#define NW   2048      // every WMMA GEMM here has N == 2048

// GEMM tiling
#define BM 64
#define BN 64
#define BK 64
#define APITCH 80      // LDS row pitch in halves (160B, multiple of 16B)

typedef __attribute__((ext_vector_type(16))) _Float16 v16h;
typedef __attribute__((ext_vector_type(8)))  _Float16 v8h;
typedef __attribute__((ext_vector_type(8)))  float    v8f;
typedef __attribute__((ext_vector_type(4)))  int      v4i;

// Async global->LDS staging (gfx1250 ASYNCcnt path), feature-gated so a
// missing builtin falls back to a register bounce instead of failing.
// Verified signature: (AS1 int4* src, AS3 int4* dst, imm offset, imm cpol)
#if defined(__HIP_DEVICE_COMPILE__) && \
    __has_builtin(__builtin_amdgcn_global_load_async_to_lds_b128) && \
    __has_builtin(__builtin_amdgcn_s_wait_asynccnt)
#define HAVE_ASYNC 1
typedef __attribute__((address_space(1))) v4i* gv4i_p;
typedef __attribute__((address_space(3))) v4i* lv4i_p;
#else
#define HAVE_ASYNC 0
#endif

__device__ __forceinline__ float sigm(float x) { return 1.0f / (1.0f + __expf(-x)); }

// ---------------------------------------------------------------------------
// Fragment load: builds the 16-bit 16x32 WMMA operand from an LDS tile
// laid out row-major [row][APITCH halves]; kk selects the 32-wide K subtile.
//   lanes  0-15: row = base+lane,    elems0..7 = K0..7,   elems8..15 = K16..23
//   lanes 16-31: row = base+lane-16, elems0..7 = K8..15,  elems8..15 = K24..31
// ---------------------------------------------------------------------------
__device__ __forceinline__ v16h load_frag(const _Float16* lds, int rowBase, int kk, int lane) {
    int halfSel = lane >> 4;             // 0 or 1
    int r = rowBase + (lane & 15);
    const _Float16* p = lds + r * APITCH + kk + halfSel * 8;
    v8h lo = *(const v8h*)(p);           // K 0..7   or 8..15   (+kk)
    v8h hi = *(const v8h*)(p + 16);      // K 16..23 or 24..31  (+kk)
    v16h out;
#pragma unroll
    for (int i = 0; i < 8; ++i) { out[i] = lo[i]; out[i + 8] = hi[i]; }
    return out;
}

// Stage one 64x64-half tile (rows r0.., cols k0..) into LDS.
// 512 x 16B transfers; 4 per thread; fully coalesced; 16B-aligned both sides.
__device__ __forceinline__ void stage_tile(const _Float16* __restrict__ g, int ldg,
                                           int r0, int k0, _Float16* s, int tid) {
#pragma unroll
    for (int q = tid; q < 512; q += 128) {
        int row = q >> 3, c8 = (q & 7) << 3;
        const _Float16* src = g + (size_t)(r0 + row) * ldg + k0 + c8;
        _Float16* dst = s + row * APITCH + c8;
#if HAVE_ASYNC
        __builtin_amdgcn_global_load_async_to_lds_b128(
            (gv4i_p)src, (lv4i_p)dst, 0, 0);
#else
        *(uint4*)dst = *(const uint4*)src;
#endif
    }
}

// ---------------------------------------------------------------------------
// WMMA GEMM:  out[M x 2048] = A[M x K] (f16, row-major, stride lda)
//                           @ Wn[2048 x K]^T (f16, N-major, stride ldw) + bias
// Both operands are K-contiguous, so A and B stage identically (no transpose).
// MODE 0: outF = acc + bias                         (gate pre-activations)
// MODE 1: t = tanh(acc+bias); outF = t; outH = t    (idea; outH is X1[:,0:2048])
// MODE 2: s = sigmoid(acc+bias); outH = s * ideaF   (att; fused flat = idea*att)
// Block = 128 threads (4 wave32), tile 64x64, each wave owns a 64x16 strip.
// Double-buffered LDS, async staging overlapped with WMMA compute; all
// fragments for a chunk are materialized before the 8 back-to-back WMMAs so
// the ds returns overlap the matrix pipe instead of serializing on dscnt.
// ---------------------------------------------------------------------------
template <int MODE>
__global__ __launch_bounds__(128) void k_gemm(
    const _Float16* __restrict__ A, int lda,
    const _Float16* __restrict__ Wn, int ldw,
    const float* __restrict__ bias,
    float* __restrict__ outF,
    _Float16* __restrict__ outH, int ldh,
    const float* __restrict__ ideaF,
    int K)
{
    __shared__ _Float16 lA[2][BM * APITCH];
    __shared__ _Float16 lB[2][BN * APITCH];

    const int tid  = threadIdx.x;
    const int lane = tid & 31;
    const int wave = tid >> 5;
    const int n0 = blockIdx.x * BN;
    const int m0 = blockIdx.y * BM;

    v8f acc[4];
#pragma unroll
    for (int t = 0; t < 4; ++t)
#pragma unroll
        for (int i = 0; i < 8; ++i) acc[t][i] = 0.0f;

    const int NK = K / BK;
    int buf = 0;
    stage_tile(A,  lda, m0, 0, lA[0], tid);
    stage_tile(Wn, ldw, n0, 0, lB[0], tid);

    for (int kc = 0; kc < NK; ++kc) {
#if HAVE_ASYNC
        __builtin_amdgcn_s_wait_asynccnt(0);   // my tile-kc copies are done
#endif
        __syncthreads();                       // everyone's copies are done
        const int cur = buf;
        if (kc + 1 < NK) {                     // overlap next stage with compute
            buf ^= 1;
            stage_tile(A,  lda, m0, (kc + 1) * BK, lA[buf], tid);
            stage_tile(Wn, ldw, n0, (kc + 1) * BK, lB[buf], tid);
        }
        if (kc + 2 < NK) {                     // warm L2 two chunks ahead
            __builtin_prefetch(Wn + (size_t)n0 * ldw + (kc + 2) * BK, 0, 1);
            __builtin_prefetch(A + (size_t)m0 * lda + (kc + 2) * BK, 0, 1);
        }

        // ---- materialize all fragments for this chunk, then 8 WMMAs ----
        v16h bfr[2];
        v16h afr[2][4];
#pragma unroll
        for (int kk = 0; kk < 2; ++kk) {
            bfr[kk] = load_frag(lB[cur], wave * 16, kk * 32, lane);
#pragma unroll
            for (int t = 0; t < 4; ++t)
                afr[kk][t] = load_frag(lA[cur], t * 16, kk * 32, lane);
        }
#pragma unroll
        for (int kk = 0; kk < 2; ++kk)
#pragma unroll
            for (int t = 0; t < 4; ++t)
                acc[t] = __builtin_amdgcn_wmma_f32_16x16x32_f16(
                    false, afr[kk][t], false, bfr[kk], (short)0, acc[t], false, false);
        // no trailing barrier: next iteration's top barrier orders reuse,
        // and the staged buffer alternates so compute(kc) never races stage(kc+1)
    }

    // ---- epilogue (C/D layout: vgpr r -> M=r (lanes 0-15) / M=r+8 (16-31)) ----
    const int n = n0 + wave * 16 + (lane & 15);
    const int mHalf = (lane < 16) ? 0 : 8;
    const float bn = bias[n];
#pragma unroll
    for (int t = 0; t < 4; ++t) {
#pragma unroll
        for (int r = 0; r < 8; ++r) {
            int m = m0 + t * 16 + mHalf + r;
            float v = acc[t][r] + bn;
            if constexpr (MODE == 0) {
                outF[(size_t)m * NW + n] = v;
            } else if constexpr (MODE == 1) {
                float tv = tanhf(v);
                outF[(size_t)m * NW + n] = tv;
                outH[(size_t)m * ldh + n] = (_Float16)tv;
            } else {
                float sv = sigm(v);
                outH[(size_t)m * ldh + n] = (_Float16)(sv * ideaF[(size_t)m * NW + n]);
            }
        }
    }
}

// ---------------------------------------------------------------------------
// Prep kernels
// ---------------------------------------------------------------------------
// Pure cast with pitch: out[n*pitch + off + k] = (f16) in[n*Kin + k]
__global__ void k_cast_pitch(const float* __restrict__ in, _Float16* __restrict__ out,
                             int Kin, int pitch, int off, int total) {
    int i = blockIdx.x * blockDim.x + threadIdx.x;
    if (i >= total) return;
    int n = i / Kin, k = i - n * Kin;
    out[(size_t)n * pitch + off + k] = (_Float16)in[i];
}

__global__ void k_bias_sum(const float* __restrict__ a, const float* __restrict__ b,
                           float* __restrict__ o, int n) {
    int i = blockIdx.x * blockDim.x + threadIdx.x;
    if (i < n) o[i] = a[i] + b[i];
}

__global__ void k_zero_half(_Float16* p, int n) {
    int i = blockIdx.x * blockDim.x + threadIdx.x;
    if (i < n) p[i] = (_Float16)0.0f;
}
__global__ void k_zero_float(float* p, int n) {
    int i = blockIdx.x * blockDim.x + threadIdx.x;
    if (i < n) p[i] = 0.0f;
}
// X1 tail = h1 columns [2048,2560) initialised to 0 (h1_0 = 0)
__global__ void k_zero_x1tail(_Float16* X1) {
    int idx = blockIdx.x * blockDim.x + threadIdx.x;  // BSZ*RNN
    int b = idx >> 9, j = idx & (RNN - 1);
    X1[(size_t)b * 2560 + CODE + j] = (_Float16)0.0f;
}

// ---------------------------------------------------------------------------
// LSTM pointwise (torch gate order i,f,g,o in 512-wide chunks of G)
// ---------------------------------------------------------------------------
__global__ void k_lstm1(const float* __restrict__ G, float* __restrict__ c1,
                        _Float16* __restrict__ X1, _Float16* __restrict__ X2) {
    int idx = blockIdx.x * blockDim.x + threadIdx.x;  // BSZ*RNN
    int b = idx >> 9, j = idx & (RNN - 1);
    const float* g = G + (size_t)b * G4;
    float c = sigm(g[RNN + j]) * c1[idx] + sigm(g[j]) * tanhf(g[2 * RNN + j]);
    float h = sigm(g[3 * RNN + j]) * tanhf(c);
    c1[idx] = c;
    X2[(size_t)b * 1024 + j]        = (_Float16)h;  // LSTM2 input, this step
    X1[(size_t)b * 2560 + CODE + j] = (_Float16)h;  // next step's A columns
}

__global__ void k_lstm2(const float* __restrict__ G, float* __restrict__ c2,
                        float* __restrict__ H2f, _Float16* __restrict__ H2h,
                        _Float16* __restrict__ X2) {
    int idx = blockIdx.x * blockDim.x + threadIdx.x;  // BSZ*RNN
    int b = idx >> 9, j = idx & (RNN - 1);
    const float* g = G + (size_t)b * G4;
    float c = sigm(g[RNN + j]) * c2[idx] + sigm(g[j]) * tanhf(g[2 * RNN + j]);
    float h = sigm(g[3 * RNN + j]) * tanhf(c);
    c2[idx] = c;
    H2f[idx] = h;                                   // f32 for mixture heads
    H2h[idx] = (_Float16)h;                         // A-matrix for att GEMM
    X2[(size_t)b * 1024 + RNN + j] = (_Float16)h;   // next step's LSTM2 h-input
}

// ---------------------------------------------------------------------------
// Mixture heads: one wave32 per batch row; 15 dot products of length 512,
// softmax over 5 mixture logits, reparameterized sample, tanh. t == 0.0.
// ---------------------------------------------------------------------------
__global__ __launch_bounds__(256) void k_heads(
    const float* __restrict__ H2f,
    const float* __restrict__ Wm, const float* __restrict__ bm,
    const float* __restrict__ Wv, const float* __restrict__ bv,
    const float* __restrict__ Wx, const float* __restrict__ bx,
    const float* __restrict__ eps, float* __restrict__ out, int t, int S)
{
    int wave = threadIdx.x >> 5, lane = threadIdx.x & 31;
    int b = blockIdx.x * 8 + wave;
    const float* h = H2f + (size_t)b * RNN;

    float dots[15];
#pragma unroll
    for (int w = 0; w < 15; ++w) {
        const float* Wp = (w < 5) ? (Wm + (size_t)w * RNN)
                        : (w < 10) ? (Wv + (size_t)(w - 5) * RNN)
                                   : (Wx + (size_t)(w - 10) * RNN);
        float s = 0.0f;
        for (int j = lane; j < RNN; j += 32) s += h[j] * Wp[j];
#pragma unroll
        for (int off = 16; off; off >>= 1) s += __shfl_down(s, off, 32);
        dots[w] = s;  // valid on lane 0
    }

    if (lane == 0) {
        float mean[5], var[5], mix[5];
#pragma unroll
        for (int m = 0; m < 5; ++m) {
            mean[m] = dots[m]     + bm[m];
            var[m]  = dots[m + 5] + bv[m];
            mix[m]  = dots[m +10] + bx[m];
        }
        float mx = mix[0];
#pragma unroll
        for (int m = 1; m < 5; ++m) mx = fmaxf(mx, mix[m]);
        float den = 0.0f, rho[5];
#pragma unroll
        for (int m = 0; m < 5; ++m) { rho[m] = __expf(mix[m] - mx); den += rho[m]; }
        float mu = 0.0f, sg = 0.0f, inv = 1.0f / den;
#pragma unroll
        for (int m = 0; m < 5; ++m) {
            float r = rho[m] * inv;
            mu += r * mean[m];
            sg += r * __expf(var[m]);   // t == 0.0 in the reference
        }
        float e = eps[(size_t)t * BSZ + b];
        out[((size_t)b * S + t) * 2 + 0] = tanhf(mu + sg * e);
        out[((size_t)b * S + t) * 2 + 1] = 0.0f;
    }
}

// ---------------------------------------------------------------------------
// Host-side orchestration
// ---------------------------------------------------------------------------
extern "C" void kernel_launch(void* const* d_in, const int* in_sizes, int n_in,
                              void* d_out, int out_size, void* d_ws, size_t ws_size,
                              hipStream_t stream) {
    const float* latent  = (const float*)d_in[0];
    const float* eps     = (const float*)d_in[1];
    // d_in[2] = num_segments (device scalar); S recoverable from eps size:
    const int S = in_sizes[1] / BSZ;
    const float* W_unpack = (const float*)d_in[3];
    const float* b_unpack = (const float*)d_in[4];
    const float* Wih1 = (const float*)d_in[5];
    const float* Whh1 = (const float*)d_in[6];
    const float* bih1 = (const float*)d_in[7];
    const float* bhh1 = (const float*)d_in[8];
    const float* Wih2 = (const float*)d_in[9];
    const float* Whh2 = (const float*)d_in[10];
    const float* bih2 = (const float*)d_in[11];
    const float* bhh2 = (const float*)d_in[12];
    const float* W_att = (const float*)d_in[13];
    const float* b_att = (const float*)d_in[14];
    const float* W_mean = (const float*)d_in[15];
    const float* b_mean = (const float*)d_in[16];
    const float* W_var  = (const float*)d_in[17];
    const float* b_var  = (const float*)d_in[18];
    const float* W_mix  = (const float*)d_in[19];
    const float* b_mix  = (const float*)d_in[20];
    float* out = (float*)d_out;

    // -------- workspace carve-out (256B aligned slabs) --------
    char* cur = (char*)d_ws;
    auto alloc = [&](size_t bytes) -> void* {
        void* p = (void*)cur;
        cur += (bytes + 255) & ~(size_t)255;
        return p;
    };
    // Weights stay N-major [2048][K] (= original layout, just cast to f16).
    _Float16* Wn1 = (_Float16*)alloc((size_t)NW * 2560 * 2);   // [Wih1|Whh1]
    _Float16* Wn2 = (_Float16*)alloc((size_t)NW * 1024 * 2);   // [Wih2|Whh2]
    _Float16* WnA = (_Float16*)alloc((size_t)NW * RNN  * 2);   // W_att
    _Float16* WnU = (_Float16*)alloc((size_t)NW * LAT  * 2);   // W_unpack
    _Float16* Lh  = (_Float16*)alloc((size_t)BSZ * LAT * 2);   // latent f16
    _Float16* X1  = (_Float16*)alloc((size_t)BSZ * 2560 * 2);  // [idea*att | h1]
    _Float16* X2  = (_Float16*)alloc((size_t)BSZ * 1024 * 2);  // [h1 | h2]
    _Float16* H2h = (_Float16*)alloc((size_t)BSZ * RNN * 2);
    float* ideaF  = (float*)alloc((size_t)BSZ * CODE * 4);
    float* G      = (float*)alloc((size_t)BSZ * G4 * 4);
    float* c1     = (float*)alloc((size_t)BSZ * RNN * 4);
    float* c2     = (float*)alloc((size_t)BSZ * RNN * 4);
    float* H2f    = (float*)alloc((size_t)BSZ * RNN * 4);
    float* bias1  = (float*)alloc((size_t)G4 * 4);
    float* bias2  = (float*)alloc((size_t)G4 * 4);

    const int T = 256;
    auto blocks = [&](int n) { return (n + T - 1) / T; };

    // -------- one-time per launch: weight casts + state init --------
    k_cast_pitch<<<blocks(NW * 2048), T, 0, stream>>>(Wih1, Wn1, 2048, 2560, 0,    NW * 2048);
    k_cast_pitch<<<blocks(NW * 512),  T, 0, stream>>>(Whh1, Wn1, 512,  2560, 2048, NW * 512);
    k_cast_pitch<<<blocks(NW * 512),  T, 0, stream>>>(Wih2, Wn2, 512,  1024, 0,    NW * 512);
    k_cast_pitch<<<blocks(NW * 512),  T, 0, stream>>>(Whh2, Wn2, 512,  1024, 512,  NW * 512);
    k_cast_pitch<<<blocks(NW * 512),  T, 0, stream>>>(W_att, WnA, 512, 512,  0,    NW * 512);
    k_cast_pitch<<<blocks(NW * 256),  T, 0, stream>>>(W_unpack, WnU, 256, 256, 0,  NW * 256);
    k_cast_pitch<<<blocks(BSZ * LAT), T, 0, stream>>>(latent, Lh, LAT, LAT, 0, BSZ * LAT);
    k_bias_sum<<<blocks(G4), T, 0, stream>>>(bih1, bhh1, bias1, G4);
    k_bias_sum<<<blocks(G4), T, 0, stream>>>(bih2, bhh2, bias2, G4);
    k_zero_x1tail<<<blocks(BSZ * RNN), T, 0, stream>>>(X1);
    k_zero_half<<<blocks(BSZ * 1024), T, 0, stream>>>(X2, BSZ * 1024);
    k_zero_float<<<blocks(BSZ * RNN), T, 0, stream>>>(c1, BSZ * RNN);
    k_zero_float<<<blocks(BSZ * RNN), T, 0, stream>>>(c2, BSZ * RNN);

    dim3 ggrid(NW / BN, BSZ / BM);   // (32, 4)

    // idea = tanh(latent @ W_unpack^T + b); also seeds X1[:,0:2048] (att0 = 1)
    k_gemm<1><<<ggrid, 128, 0, stream>>>(Lh, LAT, WnU, LAT, b_unpack,
                                         ideaF, X1, 2560, nullptr, LAT);

    // -------- sequential scan --------
    for (int t = 0; t < S; ++t) {
        k_gemm<0><<<ggrid, 128, 0, stream>>>(X1, 2560, Wn1, 2560, bias1,
                                             G, nullptr, 0, nullptr, 2560);
        k_lstm1<<<blocks(BSZ * RNN), T, 0, stream>>>(G, c1, X1, X2);

        k_gemm<0><<<ggrid, 128, 0, stream>>>(X2, 1024, Wn2, 1024, bias2,
                                             G, nullptr, 0, nullptr, 1024);
        k_lstm2<<<blocks(BSZ * RNN), T, 0, stream>>>(G, c2, H2f, H2h, X2);

        // att = sigmoid(h2 @ W_att^T + b); fused: X1[:,0:2048] = f16(att * idea)
        k_gemm<2><<<ggrid, 128, 0, stream>>>(H2h, RNN, WnA, RNN, b_att,
                                             nullptr, X1, 2560, ideaF, RNN);

        k_heads<<<BSZ / 8, 256, 0, stream>>>(H2f, W_mean, b_mean, W_var, b_var,
                                             W_mix, b_mix, eps, out, t, S);
    }
    (void)n_in; (void)out_size; (void)ws_size;
}
